// MLP_41068477285034
// MI455X (gfx1250) — matrix operand order repaired and checked by reading, not compiled, not run on hardware
//
#include <hip/hip_runtime.h>

// ---------------------------------------------------------------------------
// Scatter-MoE MLP for MI455X (gfx1250).
//   y[t] = sum_k p[t,k] * gelu_tanh(x[t] @ w1[e(t,k)]) @ w2[e(t,k)]
// Sparse routing: 1.10 PFLOP (vs 4.4 dense).  Compute-bound => bf16 WMMA
// 16x16x32 with f32 accumulation.  Weights are pre-converted to bf16 and
// pre-shuffled into the WMMA B-operand register layout (one 32B contiguous
// run per lane per 32x16 tile), so the hot loop is 2x global_load_b128 per
// v_wmma instead of 16x strided global_load_b32 + 8x v_cvt_pk_bf16_f32.
// Pre-pass cost: 1.07GB read + 0.54GB write ~= 70us at 23.3 TB/s.
// ---------------------------------------------------------------------------

#define TT 8192
#define DD 2048
#define HH 8192
#define EE 8
#define TM 16                    // tokens per workgroup tile (one WMMA M tile)
#define KH 64                    // H-chunk per pipeline step
#define TILES_PER_E (TT / TM)    // 512 (worst case: all tokens on one expert)

typedef __attribute__((ext_vector_type(16))) __bf16       v16bf;
typedef __attribute__((ext_vector_type(8)))  float        v8f;
typedef __attribute__((ext_vector_type(4)))  float        v4f;
typedef __attribute__((ext_vector_type(4)))  unsigned int v4u;

__device__ __forceinline__ float gelu_tanh(float x) {
  // jax.nn.gelu default (approximate=True)
  float x3 = x * x * x;
  float u  = 0.7978845608028654f * (x + 0.044715f * x3);
  return 0.5f * x * (1.0f + tanhf(u));
}

// Fallback: build a 32x16 bf16 B operand from row-major f32 W (leading dim
// ldw).  B layout used here (16-bit, K=32, contiguous K): lanes 0-15 hold K=0..15 (2/VGPR),
// lanes 16-31 hold K=16..31; N = lane%16.
__device__ __forceinline__ v16bf load_b_f32(const float* __restrict__ W,
                                            int krow, int ncol, int ldw,
                                            int lane) {
  const int n   = lane & 15;
  const int kof = (lane & 16) ? 16 : 0;
  const float* p = W + (size_t)(krow + kof) * (size_t)ldw + (size_t)(ncol + n);
  v16bf b;
#pragma unroll
  for (int v = 0; v < 8; ++v) {
    float f0 = p[0];
    float f1 = p[ldw];
    b[2 * v]     = (__bf16)f0;
    b[2 * v + 1] = (__bf16)f1;
    p += (size_t)(2 * ldw);
  }
  return b;
}

// ---------------------------------------------------------------------------
// Pass 0: zero output (harness poisons d_out) and per-expert counters.
// ---------------------------------------------------------------------------
__global__ void moe_init_kernel(float* __restrict__ y, int* __restrict__ counts) {
  size_t i = (size_t)blockIdx.x * blockDim.x + threadIdx.x;
  size_t n = (size_t)TT * DD;
  for (size_t k = i; k < n; k += (size_t)gridDim.x * blockDim.x) y[k] = 0.0f;
  if (i < EE) counts[i] = 0;
}

// ---------------------------------------------------------------------------
// Pass 1: bin tokens per expert (token id + summed gate prob).
// ---------------------------------------------------------------------------
__global__ void moe_route_kernel(const long long* __restrict__ idx,
                                 const float* __restrict__ p,
                                 int* __restrict__ counts,
                                 int* __restrict__ tok_list,
                                 float* __restrict__ gate_list) {
  int t = blockIdx.x * blockDim.x + threadIdx.x;
  if (t >= TT) return;
  int   e0 = (int)idx[2 * t];
  int   e1 = (int)idx[2 * t + 1];
  float p0 = p[2 * t];
  float p1 = p[2 * t + 1];
  if (e0 == e1) {
    int pos = atomicAdd(&counts[e0], 1);
    tok_list[e0 * TT + pos]  = t;
    gate_list[e0 * TT + pos] = p0 + p1;
  } else {
    int pos0 = atomicAdd(&counts[e0], 1);
    tok_list[e0 * TT + pos0]  = t;
    gate_list[e0 * TT + pos0] = p0;
    int pos1 = atomicAdd(&counts[e1], 1);
    tok_list[e1 * TT + pos1]  = t;
    gate_list[e1 * TT + pos1] = p1;
  }
}

// ---------------------------------------------------------------------------
// Pass 1b: convert row-major f32 weights [Kr x Nc] into bf16 tiles in the
// WMMA B-operand register layout.  Tile (kt, nt) covers K rows kt*32..+31 and
// N cols nt*16..+15.  Within a tile (512 bf16 = 256 dwords):
//   dword[lane*8 + v] = pack(W[kt*32+kof+2v][nt*16+n], W[kt*32+kof+2v+1][...])
// with n = lane&15, kof = (lane&16)?16:0.  Main loop then loads 32 contiguous
// bytes per lane.  Reads are coalesced along n (consecutive threads).
// ---------------------------------------------------------------------------
__global__ void moe_shuffle_kernel(const float* __restrict__ W,
                                   unsigned* __restrict__ out,
                                   int Kr, int Nc) {
  const size_t total   = (size_t)Kr * (size_t)Nc / 2;   // output dwords
  const int    ntilesn = Nc >> 4;
  const size_t stride  = (size_t)gridDim.x * blockDim.x;
  for (size_t d = (size_t)blockIdx.x * blockDim.x + threadIdx.x; d < total;
       d += stride) {
    size_t tile = d >> 8;                 // 256 dwords per tile
    int    dw   = (int)(d & 255);
    int    v    = dw >> 5;                // 0..7 (dword within lane run)
    int    lane = dw & 31;
    int    kof  = (lane & 16) ? 16 : 0;
    int    n    = lane & 15;
    size_t kt   = tile / ntilesn;
    int    nt   = (int)(tile % ntilesn);
    size_t k    = kt * 32 + kof + 2 * v;
    const float* p = W + k * (size_t)Nc + (size_t)(nt * 16 + n);
    __bf16 lo = (__bf16)p[0];
    __bf16 hi = (__bf16)p[Nc];
    unsigned u = (unsigned)__builtin_bit_cast(unsigned short, lo) |
                 ((unsigned)__builtin_bit_cast(unsigned short, hi) << 16);
    out[tile * 256 + (size_t)(lane * 8 + v)] = u;
  }
}

// ---------------------------------------------------------------------------
// Pass 2: fused expert MLP.  One WG = 16 tokens of one expert; 8 waves.
//  - wave w permanently holds bf16 A-operands for D-slice [w*256, w*256+256)
//  - per 64-wide H chunk: GEMM1 partials -> LDS reduce + gelu*gate -> bf16 hA
//    -> GEMM2 into 128 f32 accumulator VGPRs (16 N-tiles of the D output)
//  - SHUF path: B operands are single 32B vector loads from shuffled weights
//  - f32 atomics into y (a token's two experts live in different WGs)
// ---------------------------------------------------------------------------
template <bool SHUF>
__global__ void __launch_bounds__(256)
moe_mlp_kernel(const float* __restrict__ x,
               const float* __restrict__ w1,
               const float* __restrict__ w2,
               const __bf16* __restrict__ w1s,
               const __bf16* __restrict__ w2s,
               const int* __restrict__ counts,
               const int* __restrict__ tok_list,
               const float* __restrict__ gate_list,
               float* __restrict__ y) {
  __shared__ float  part[8][TM][KH];     // 32 KB cross-wave GEMM1 partials
  __shared__ __bf16 hA[TM][KH + 8];      // gated gelu(h) chunk, padded rows
  __shared__ int    s_tok[TM];
  __shared__ float  s_gate[TM];

  const int e    = blockIdx.x / TILES_PER_E;
  const int tile = blockIdx.x % TILES_PER_E;
  const int cnt  = counts[e];
  const int base = tile * TM;
  if (base >= cnt) return;               // uniform: EXEC stays all-ones

  const int tid  = threadIdx.x;
  const int wave = tid >> 5;
  const int lane = tid & 31;

  if (tid < TM) {
    int r  = base + tid;
    int rc = (r < cnt) ? r : (cnt - 1);  // pad with dup token, gate = 0
    s_tok[tid]  = tok_list[e * TT + rc];
    s_gate[tid] = (r < cnt) ? gate_list[e * TT + r] : 0.0f;
  }
  __syncthreads();

  const float*  w1e  = w1 + (size_t)e * DD * HH;
  const float*  w2e  = w2 + (size_t)e * HH * DD;
  const __bf16* w1se = SHUF ? (w1s + (size_t)e * DD * HH) : nullptr;
  const __bf16* w2se = SHUF ? (w2s + (size_t)e * HH * DD) : nullptr;

  // --- preload this wave's x A-operands (contiguous-K 16-bit A layout, K depth 256) ---
  const int dbase = wave * 256;
  const int m     = lane & 15;
  const int kofA  = (lane & 16) ? 16 : 0;  // lanes 16-31 hold K 16-31
  const float* xr = x + (size_t)s_tok[m] * DD + dbase;

  v16bf a_ops[8];
#pragma unroll
  for (int ks = 0; ks < 8; ++ks) {
    const float* p0 = xr + ks * 32 + kofA;
    v4f f0 = *(const v4f*)(p0);
    v4f f1 = *(const v4f*)(p0 + 4);
    v4f f2 = *(const v4f*)(p0 + 8);
    v4f f3 = *(const v4f*)(p0 + 12);
    v16bf a;
#pragma unroll
    for (int j = 0; j < 4; ++j) {
      a[j]      = (__bf16)f0[j];
      a[4 + j]  = (__bf16)f1[j];
      a[8 + j]  = (__bf16)f2[j];
      a[12 + j] = (__bf16)f3[j];
    }
    a_ops[ks] = a;
  }

  v8f accY[16];
#pragma unroll
  for (int i = 0; i < 16; ++i)
#pragma unroll
    for (int r = 0; r < 8; ++r) accY[i][r] = 0.0f;

  const int n16 = lane & 15;
  const int mh  = (lane & 16) ? 8 : 0;

  for (int c = 0; c < HH / KH; ++c) {
    const int hbase = c * KH;

    if (SHUF && c + 1 < HH / KH) {
      // prefetch next chunk's weight tiles (global_prefetch_b8)
      size_t t1 = ((size_t)(wave * 8) * (HH / 16) + ((hbase + KH) >> 4));
      size_t t2 = ((size_t)((hbase + KH) >> 5) * (DD / 16) + wave * 16);
      __builtin_prefetch((const void*)(w1se + t1 * 512), 0, 1);
      __builtin_prefetch((const void*)(w2se + t2 * 512), 0, 1);
    }

    // ---- GEMM1: partials over this wave's 256-deep D slice ----
#pragma unroll
    for (int nt = 0; nt < 4; ++nt) {
      v8f acc1;
#pragma unroll
      for (int r = 0; r < 8; ++r) acc1[r] = 0.0f;
#pragma unroll
      for (int ks = 0; ks < 8; ++ks) {
        v16bf b;
        if (SHUF) {
          size_t t1 = (size_t)(wave * 8 + ks) * (HH / 16) +
                      (size_t)((hbase >> 4) + nt);
          b = *(const v16bf*)(w1se + t1 * 512 + lane * 16);
        } else {
          b = load_b_f32(w1e, dbase + ks * 32, hbase + nt * 16, HH, lane);
        }
        acc1 = __builtin_amdgcn_wmma_f32_16x16x32_bf16(
            false, a_ops[ks], false, b, (short)0, acc1, false, false);
      }
#pragma unroll
      for (int r = 0; r < 8; ++r)
        part[wave][mh + r][nt * 16 + n16] = acc1[r];
    }
    __syncthreads();

    // ---- reduce across 8 waves, gelu, fold gate, convert to bf16 ----
#pragma unroll
    for (int j = 0; j < 4; ++j) {
      int   i  = tid * 4 + j;    // 0..1023 over [16][64]
      int   mm = i >> 6;
      int   nn = i & 63;
      float s  = 0.0f;
#pragma unroll
      for (int w = 0; w < 8; ++w) s += part[w][mm][nn];
      s = gelu_tanh(s) * s_gate[mm];
      hA[mm][nn] = (__bf16)s;
    }
    __syncthreads();

    // ---- GEMM2: accY += hA @ w2 chunk (wave's 256-wide output slice) ----
    v16bf ah[2];
#pragma unroll
    for (int ks = 0; ks < 2; ++ks) {
      union { v16bf v; v4u q[2]; } u;
      u.q[0] = *(const v4u*)&hA[m][ks * 32 + kofA];        // K 0-7 / 16-23
      u.q[1] = *(const v4u*)&hA[m][ks * 32 + 8 + kofA];    // K 8-15 / 24-31
      ah[ks] = u.v;
    }
#pragma unroll
    for (int nt = 0; nt < 16; ++nt) {
#pragma unroll
      for (int ks = 0; ks < 2; ++ks) {
        v16bf b;
        if (SHUF) {
          size_t t2 = (size_t)((hbase >> 5) + ks) * (DD / 16) +
                      (size_t)(wave * 16 + nt);
          b = *(const v16bf*)(w2se + t2 * 512 + lane * 16);
        } else {
          b = load_b_f32(w2e, hbase + ks * 32, dbase + nt * 16, DD, lane);
        }
        accY[nt] = __builtin_amdgcn_wmma_f32_16x16x32_bf16(
            false, ah[ks], false, b, (short)0, accY[nt], false, false);
      }
    }
    __syncthreads();  // protect part[] before next chunk's GEMM1 writes
  }

  // ---- scatter-accumulate into y ----
#pragma unroll
  for (int nt = 0; nt < 16; ++nt) {
    int col = dbase + nt * 16 + n16;
#pragma unroll
    for (int r = 0; r < 8; ++r) {
      int mm = mh + r;
      atomicAdd(&y[(size_t)s_tok[mm] * DD + col], accY[nt][r]);
    }
  }
}

// ---------------------------------------------------------------------------
extern "C" void kernel_launch(void* const* d_in, const int* in_sizes, int n_in,
                              void* d_out, int out_size, void* d_ws,
                              size_t ws_size, hipStream_t stream) {
  (void)in_sizes; (void)n_in; (void)out_size;

  const float*     x  = (const float*)d_in[0];
  const float*     ep = (const float*)d_in[1];
  const long long* ei = (const long long*)d_in[2];  // jnp.int64 in reference
  const float*     w1 = (const float*)d_in[3];
  const float*     w2 = (const float*)d_in[4];
  float*           y  = (float*)d_out;

  // workspace layout:
  //   [counts: E ints][tok_list: E*T ints][gate_list: E*T f32]  (< 1 MB)
  //   @1MB: w1 shuffled bf16 (256 MB), then w2 shuffled bf16 (256 MB)
  int*   counts    = (int*)d_ws;
  int*   tok_list  = (int*)((char*)d_ws + 256);
  float* gate_list = (float*)((char*)d_ws + 256 + sizeof(int) * EE * TT);

  const size_t welems   = (size_t)EE * DD * HH;            // per matrix
  const size_t shuf_off = (size_t)1 << 20;
  const size_t need     = shuf_off + 2 * welems * sizeof(__bf16);
  __bf16* w1s = (__bf16*)((char*)d_ws + shuf_off);
  __bf16* w2s = (__bf16*)((char*)d_ws + shuf_off + welems * sizeof(__bf16));

  moe_init_kernel<<<2048, 256, 0, stream>>>(y, counts);
  moe_route_kernel<<<(TT + 255) / 256, 256, 0, stream>>>(ei, ep, counts,
                                                         tok_list, gate_list);
  if (ws_size >= need) {
    moe_shuffle_kernel<<<4096, 256, 0, stream>>>(w1, (unsigned*)w1s,
                                                 EE * DD, HH);
    moe_shuffle_kernel<<<4096, 256, 0, stream>>>(w2, (unsigned*)w2s,
                                                 EE * HH, DD);
    moe_mlp_kernel<true><<<EE * TILES_PER_E, 256, 0, stream>>>(
        x, w1, w2, w1s, w2s, counts, tok_list, gate_list, y);
  } else {
    moe_mlp_kernel<false><<<EE * TILES_PER_E, 256, 0, stream>>>(
        x, w1, w2, nullptr, nullptr, counts, tok_list, gate_list, y);
  }
}
